// DynamicAttention3_40827959116536
// MI455X (gfx1250) — matrix operand order repaired and checked by reading, not compiled, not run on hardware
//
#include <hip/hip_runtime.h>
#include <math.h>

// MI455X / gfx1250: wave32, WMMA 16x16x32 bf16 (f32 accum).
typedef __attribute__((ext_vector_type(16))) __bf16 v16bf;
typedef __attribute__((ext_vector_type(8)))  __bf16 v8bf;
typedef __attribute__((ext_vector_type(4)))  __bf16 v4bf;
typedef __attribute__((ext_vector_type(8)))  float  v8f;

// ---- WMMA fragment loaders (layouts per CDNA5 ISA §7.12.2) ----

// A (16x32 bf16, row-major LDS tile):
// lanes 0-15: M=lane, K={0..7,16..23}; lanes 16-31: M=lane-16, K={8..15,24..31}
__device__ __forceinline__ v16bf frag_A_16x32(const __bf16* base, int stride) {
  int lane = threadIdx.x & 31;
  int r = lane & 15;
  int kof = (lane & 16) ? 8 : 0;
  const __bf16* p = base + r * stride + kof;
  v8bf lo = *reinterpret_cast<const v8bf*>(p);
  v8bf hi = *reinterpret_cast<const v8bf*>(p + 16);
  v16bf f;
#pragma unroll
  for (int i = 0; i < 8; ++i) { f[i] = lo[i]; f[i + 8] = hi[i]; }
  return f;
}

// B (32x16 bf16) from transposed row-major [N][K] tile:
// lane n holds column N=n; lanes 0-15: K={0..7,16..23}, lanes 16-31: K={8..15,24..31}
__device__ __forceinline__ v16bf frag_B_32x16(const __bf16* base, int stride) {
  int lane = threadIdx.x & 31;
  int n = lane & 15;
  int kof = (lane & 16) ? 8 : 0;
  const __bf16* p = base + n * stride + kof;
  v8bf lo = *reinterpret_cast<const v8bf*>(p);
  v8bf hi = *reinterpret_cast<const v8bf*>(p + 16);
  v16bf f;
#pragma unroll
  for (int i = 0; i < 8; ++i) { f[i] = lo[i]; f[i + 8] = hi[i]; }
  return f;
}

// ---- bf16-WMMA GEMM: C[M,N] = act(A[M,K] @ W[K,N] + bias) ----
// Tile 128(M) x 64(N), K-step 32. 256 threads = 8 waves; wave w owns a 16x64
// strip: one A fragment reused across 4 B fragments -> 4 WMMA per wave/iter.
// rpo/rpi: padded rows per batch (out) / real rows per batch (in); rows with
// (m % rpo) >= rpi read A as zero (the s+1/t+1 pad row). act: 1 = tanh.
__global__ __launch_bounds__(256)
void gemm_bf16(const float* __restrict__ A, const float* __restrict__ W,
               const float* __restrict__ bias, float* __restrict__ C,
               int M, int N, int K, int rpo, int rpi, int act) {
  __shared__ __attribute__((aligned(16))) __bf16 A_s[128 * 32];  // [m][k]
  __shared__ __attribute__((aligned(16))) __bf16 Bt_s[64 * 32];  // [n][k]
  const int tid = threadIdx.x, lane = tid & 31, w = tid >> 5;
  const int bN = blockIdx.x * 64, bM = blockIdx.y * 128;

  v8f zero8 = {0.f,0.f,0.f,0.f,0.f,0.f,0.f,0.f};
  v8f acc[4] = {zero8, zero8, zero8, zero8};

  // A stage: 2 threads per row, 16 contiguous k each (4 x float4 -> 16 bf16)
  const int rA = tid >> 1, cA = (tid & 1) * 16;
  // W stage: thread handles k=kB, 8 contiguous n (2 x float4), transposed store
  const int kB = tid >> 3, nB = (tid & 7) * 8;

  // resolve pad mapping once (row is fixed across k-loop)
  int mrow = bM + rA;
  bool zrow = false;
  if (rpo) { int bb = mrow / rpo; int r = mrow - bb * rpo; if (r >= rpi) zrow = true; mrow = bb * rpi + r; }
  const float* aRow = A + (size_t)mrow * K + cA;

  for (int k0 = 0; k0 < K; k0 += 32) {
    { // stage A tile (128x32) f32 -> bf16, wide loads under ONE branch
      float4 a0 = {0.f,0.f,0.f,0.f}, a1 = a0, a2 = a0, a3 = a0;
      if (!zrow) {
        const float4* ap = reinterpret_cast<const float4*>(aRow + k0);
        a0 = ap[0]; a1 = ap[1]; a2 = ap[2]; a3 = ap[3];
        if (k0 + 32 < K) __builtin_prefetch(aRow + k0 + 32, 0, 1);  // global_prefetch_b8
      }
      v8bf p0, p1;
      p0[0]=(__bf16)a0.x; p0[1]=(__bf16)a0.y; p0[2]=(__bf16)a0.z; p0[3]=(__bf16)a0.w;
      p0[4]=(__bf16)a1.x; p0[5]=(__bf16)a1.y; p0[6]=(__bf16)a1.z; p0[7]=(__bf16)a1.w;
      p1[0]=(__bf16)a2.x; p1[1]=(__bf16)a2.y; p1[2]=(__bf16)a2.z; p1[3]=(__bf16)a2.w;
      p1[4]=(__bf16)a3.x; p1[5]=(__bf16)a3.y; p1[6]=(__bf16)a3.z; p1[7]=(__bf16)a3.w;
      *reinterpret_cast<v8bf*>(&A_s[rA * 32 + cA])     = p0;
      *reinterpret_cast<v8bf*>(&A_s[rA * 32 + cA + 8]) = p1;
    }
    { // stage W tile transposed -> [n][k]
      const float* wp = W + (size_t)(k0 + kB) * N + bN + nB;
      if (k0 + 32 < K) __builtin_prefetch(wp + (size_t)32 * N, 0, 1);
      float4 w0 = reinterpret_cast<const float4*>(wp)[0];
      float4 w1 = reinterpret_cast<const float4*>(wp)[1];
      Bt_s[(nB + 0) * 32 + kB] = (__bf16)w0.x;
      Bt_s[(nB + 1) * 32 + kB] = (__bf16)w0.y;
      Bt_s[(nB + 2) * 32 + kB] = (__bf16)w0.z;
      Bt_s[(nB + 3) * 32 + kB] = (__bf16)w0.w;
      Bt_s[(nB + 4) * 32 + kB] = (__bf16)w1.x;
      Bt_s[(nB + 5) * 32 + kB] = (__bf16)w1.y;
      Bt_s[(nB + 6) * 32 + kB] = (__bf16)w1.z;
      Bt_s[(nB + 7) * 32 + kB] = (__bf16)w1.w;
    }
    __syncthreads();
    v16bf af = frag_A_16x32(&A_s[(16 * w) * 32], 32);
#pragma unroll
    for (int j = 0; j < 4; ++j) {
      v16bf bf = frag_B_32x16(&Bt_s[(16 * j) * 32], 32);
      acc[j] = __builtin_amdgcn_wmma_f32_16x16x32_bf16(false, af, false, bf, (short)0, acc[j], false, false);
    }
    __syncthreads();
  }

  // epilogue: C/D layout => row = 16*w + r + 8*(lane>=16), col = 16*j + (lane&15)
  const int rowHalf = (lane & 16) ? 8 : 0;
#pragma unroll
  for (int j = 0; j < 4; ++j) {
    int col = bN + 16 * j + (lane & 15);
    float bv = bias[col];
#pragma unroll
    for (int r = 0; r < 8; ++r) {
      int row = bM + 16 * w + r + rowHalf;
      float v = acc[j][r] + bv;
      if (act) v = tanhf(v);
      C[(size_t)row * N + col] = v;
    }
  }
}

// ---- Fused trilinear attention per (b,l): scores via WMMA, joint softmax,
//      marginal contexts, and assembly of X = [query | s_ctx | t_ctx]. ----
#define SP 144   // 129 (s+1 / t+1) padded to 9 tiles of 16
#define HD 512

__global__ __launch_bounds__(288)
void attn_kernel(const float* __restrict__ qbuf, const float* __restrict__ skey,
                 const float* __restrict__ tkey, const float* __restrict__ query,
                 float* __restrict__ X) {
  __shared__ __attribute__((aligned(16))) float  q_lds[HD];
  __shared__ __attribute__((aligned(16))) __bf16 A_lds[SP * 32];  // s_key * q_l / sqrt(H)
  __shared__ __attribute__((aligned(16))) __bf16 B_lds[SP * 32];  // t_key chunk
  __shared__ float ws_lds[SP];
  __shared__ float wt_lds[SP];
  __shared__ float wmax[9];

  const int tid = threadIdx.x, lane = tid & 31, w = tid >> 5;  // 9 waves: s-strip id
  const int bl = blockIdx.x;          // b*L + l
  const int b  = bl >> 8;             // L = 256

  const float scale = 0.04419417382415922f;  // 1/sqrt(512)
  for (int i = tid; i < HD; i += 288) q_lds[i] = qbuf[(size_t)bl * HD + i] * scale;

  v8f zero8 = {0.f,0.f,0.f,0.f,0.f,0.f,0.f,0.f};
  v8f acc[9];
#pragma unroll
  for (int j = 0; j < 9; ++j) acc[j] = zero8;
  __syncthreads();

  const float* sb = skey + (size_t)b * SP * HD;
  const float* tb = tkey + (size_t)b * SP * HD;

  // score GEMM: (diag(q_l) * s_key) @ t_key^T, K chunks of 32
  for (int k0 = 0; k0 < HD; k0 += 32) {
    // vectorized stage: each thread moves 4 consecutive elements (float4 in,
    // 4x bf16 / 8B out); SP*32/4 = 1152 quads, 288 threads -> 4 iterations
    for (int q4 = tid; q4 < SP * 8; q4 += 288) {
      int r = q4 >> 3, c = (q4 & 7) * 4;
      float4 sv = *reinterpret_cast<const float4*>(&sb[(size_t)r * HD + k0 + c]);
      float4 tv = *reinterpret_cast<const float4*>(&tb[(size_t)r * HD + k0 + c]);
      v4bf sa, ta;
      sa[0] = (__bf16)(sv.x * q_lds[k0 + c + 0]);
      sa[1] = (__bf16)(sv.y * q_lds[k0 + c + 1]);
      sa[2] = (__bf16)(sv.z * q_lds[k0 + c + 2]);
      sa[3] = (__bf16)(sv.w * q_lds[k0 + c + 3]);
      ta[0] = (__bf16)tv.x; ta[1] = (__bf16)tv.y; ta[2] = (__bf16)tv.z; ta[3] = (__bf16)tv.w;
      *reinterpret_cast<v4bf*>(&A_lds[r * 32 + c]) = sa;
      *reinterpret_cast<v4bf*>(&B_lds[r * 32 + c]) = ta;
    }
    __syncthreads();
    v16bf af = frag_A_16x32(&A_lds[(16 * w) * 32], 32);
#pragma unroll
    for (int j = 0; j < 9; ++j) {
      v16bf bfr = frag_B_32x16(&B_lds[(16 * j) * 32], 32);
      acc[j] = __builtin_amdgcn_wmma_f32_16x16x32_bf16(false, af, false, bfr, (short)0, acc[j], false, false);
    }
    __syncthreads();
  }

  // mask pad rows/cols (valid s,t in [0,128]) and wave-local max
  const int sHalf = (lane & 16) ? 8 : 0;
  float m = -3.0e38f;
#pragma unroll
  for (int j = 0; j < 9; ++j) {
    int t = 16 * j + (lane & 15);
#pragma unroll
    for (int r = 0; r < 8; ++r) {
      int s = 16 * w + r + sHalf;
      if (s > 128 || t > 128) acc[j][r] = -3.0e38f;
      m = fmaxf(m, acc[j][r]);
    }
  }
#pragma unroll
  for (int o = 16; o > 0; o >>= 1) m = fmaxf(m, __shfl_xor(m, o));
  if (lane == 0) wmax[w] = m;
  for (int i = tid; i < SP; i += 288) { ws_lds[i] = 0.f; wt_lds[i] = 0.f; }
  __syncthreads();
  float gm = wmax[0];
#pragma unroll
  for (int i = 1; i < 9; ++i) gm = fmaxf(gm, wmax[i]);

  // exp + column (t) marginals
#pragma unroll
  for (int j = 0; j < 9; ++j) {
    float cp = 0.f;
#pragma unroll
    for (int r = 0; r < 8; ++r) {
      float e = __expf(acc[j][r] - gm);
      acc[j][r] = e;
      cp += e;
    }
    cp += __shfl_xor(cp, 16);                              // merge the two M-halves
    if (lane < 16) atomicAdd(&wt_lds[16 * j + lane], cp);  // ds_add_f32 across waves
  }
  // row (s) marginals: reduce across the 16 lanes of each half
#pragma unroll
  for (int r = 0; r < 8; ++r) {
    float rp = 0.f;
#pragma unroll
    for (int j = 0; j < 9; ++j) rp += acc[j][r];
#pragma unroll
    for (int o = 1; o < 16; o <<= 1) rp += __shfl_xor(rp, o);
    if ((lane & 15) == 0) ws_lds[16 * w + r + sHalf] = rp;
  }
  __syncthreads();

  float Z = 0.f;
  for (int i = 0; i < SP; ++i) Z += ws_lds[i];
  float invZ = 1.f / Z;

  // rank-1 contexts + assemble X = [query | s_ctx | t_ctx] (loads coalesced over k)
  for (int k = tid; k < HD; k += 288) {
    float sa = 0.f, ta = 0.f;
    for (int r = 0; r < SP; ++r) {
      sa += ws_lds[r] * sb[(size_t)r * HD + k];
      ta += wt_lds[r] * tb[(size_t)r * HD + k];
    }
    size_t xo = (size_t)bl * 1536;
    X[xo + k]        = query[(size_t)bl * HD + k];
    X[xo + 512 + k]  = sa * invZ;
    X[xo + 1024 + k] = ta * invZ;
  }
}

extern "C" void kernel_launch(void* const* d_in, const int* in_sizes, int n_in,
                              void* d_out, int out_size, void* d_ws, size_t ws_size,
                              hipStream_t stream) {
  (void)in_sizes; (void)n_in; (void)out_size; (void)ws_size;
  const float* query = (const float*)d_in[0];
  const float* src   = (const float*)d_in[1];
  const float* trg   = (const float*)d_in[2];
  const float* Wq    = (const float*)d_in[3];
  const float* bq    = (const float*)d_in[4];
  const float* Ws    = (const float*)d_in[5];
  const float* bs    = (const float*)d_in[6];
  const float* Wt    = (const float*)d_in[7];
  const float* bt    = (const float*)d_in[8];
  const float* Wo    = (const float*)d_in[9];
  const float* bo    = (const float*)d_in[10];
  float* out = (float*)d_out;

  // workspace (f32): q[4096*512] | s_key[2304*512] | t_key[2304*512] | X[4096*1536]
  float* qbuf = (float*)d_ws;
  float* skey = qbuf + (size_t)4096 * 512;
  float* tkey = skey + (size_t)2304 * 512;
  float* Xbuf = tkey + (size_t)2304 * 512;

  dim3 blk(256);
  // q = query @ Wq + bq                      (4096 x 512 x 512)
  gemm_bf16<<<dim3(8, 32), blk, 0, stream>>>(query, Wq, bq, qbuf, 4096, 512, 512, 0, 0, 0);
  // s_key = [src; 0-pad] @ Ws + bs           (16 batches x 144(129 used) rows)
  gemm_bf16<<<dim3(8, 18), blk, 0, stream>>>(src, Ws, bs, skey, 2304, 512, 512, 144, 128, 0);
  // t_key = [trg; 0-pad] @ Wt + bt
  gemm_bf16<<<dim3(8, 18), blk, 0, stream>>>(trg, Wt, bt, tkey, 2304, 512, 512, 144, 128, 0);
  // fused trilinear scores + joint softmax + marginal contexts + concat
  attn_kernel<<<dim3(4096), dim3(288), 0, stream>>>(qbuf, skey, tkey, query, Xbuf);
  // out = tanh(X @ Wo + bo)                  (4096 x 512 x 1536)
  gemm_bf16<<<dim3(8, 32), blk, 0, stream>>>(Xbuf, Wo, bo, out, 4096, 512, 1536, 0, 0, 1);
}
